// DGCNN_44478681317800
// MI455X (gfx1250) — matrix operand order repaired and checked by reading, not compiled, not run on hardware
//
#include <hip/hip_runtime.h>
#include <hip/hip_bf16.h>

// ---------------------------------------------------------------------------
// DGCNN forward for MI455X (gfx1250, wave32).
// All GEMM-like work goes through v_wmma_f32_16x16x32_bf16.
// Features live as bf16 in point-major layout fb[b][n][c] so that both A and
// B WMMA fragments read aligned 32-bit channel pairs.
// K-chunk counts are template parameters so the WMMA loops are straight-line.
// ---------------------------------------------------------------------------

typedef __attribute__((ext_vector_type(16))) __bf16        v16bf;
typedef __attribute__((ext_vector_type(8)))  float         v8f;
typedef __attribute__((ext_vector_type(8)))  unsigned int  v8u;

#define BATCH 16
#define NPTS  1024
#define KNN   20

// ------------------------- scalar bf16 helpers -----------------------------
__device__ __forceinline__ unsigned short f2bf(float f) {
  unsigned u = __float_as_uint(f);
  unsigned r = (u + 0x7FFFu + ((u >> 16) & 1u)) >> 16;   // RNE
  return (unsigned short)r;
}
__device__ __forceinline__ float bf2f(unsigned short h) {
  return __uint_as_float(((unsigned)h) << 16);
}

// monotone float<->uint key for integer max atomics (handles negatives)
__device__ __forceinline__ unsigned fkey(float f) {
  unsigned u = __float_as_uint(f);
  return (u & 0x80000000u) ? ~u : (u | 0x80000000u);
}
__device__ __forceinline__ float finv(unsigned k) {
  return __uint_as_float((k & 0x80000000u) ? (k & 0x7FFFFFFFu) : ~k);
}

// ------------------------- WMMA fragment loaders ---------------------------
// A-matrix 16x32 bf16 (ISA 7.12.2): lane m=lane&15, h=lane>>4,
// vgpr v holds K = (v>>2)*16 + h*8 + (v&3)*2  (+1).  base -> element [0][0],
// row stride rs (bf16 elements), rows are M, channels contiguous.
__device__ __forceinline__ v16bf load_frag_a(const unsigned short* base, int rs, int lane) {
  int m = lane & 15, h = lane >> 4;
  const unsigned short* rowp = base + (size_t)m * rs;
  v8u u;
#pragma unroll
  for (int v = 0; v < 8; ++v) {
    int k = ((v >> 2) << 4) + (h << 3) + ((v & 3) << 1);
    u[v] = *(const unsigned int*)(rowp + k);
  }
  return __builtin_bit_cast(v16bf, u);
}

// B-matrix 32x16 bf16: lane n=lane&15, h=lane>>4, vgpr v holds K = (v>>2)*16 + h*8 + (v&3)*2 (+1).
// Memory is column(point)-major: column stride rs, channels contiguous.
__device__ __forceinline__ v16bf load_frag_b(const unsigned short* base, int rs, int lane) {
  int n = lane & 15, h = lane >> 4;
  const unsigned short* colp = base + (size_t)n * rs;
  v8u u;
#pragma unroll
  for (int v = 0; v < 8; ++v) {
    int k = ((v >> 2) << 4) + (h << 3) + ((v & 3) << 1);
    u[v] = *(const unsigned int*)(colp + k);
  }
  return __builtin_bit_cast(v16bf, u);
}

__device__ __forceinline__ v8f wmma_bf16(v16bf a, v16bf b, v8f c) {
  return __builtin_amdgcn_wmma_f32_16x16x32_bf16(false, a, false, b, (short)0, c, false, false);
}

// ------------------------- cast / prep kernels -----------------------------
// x (B,3,N) f32  ->  xb0 (B,N,32) bf16 zero padded
__global__ void cast_x0_kernel(const float* __restrict__ x, unsigned short* __restrict__ xb0) {
  int t = blockIdx.x * blockDim.x + threadIdx.x;
  if (t >= BATCH * NPTS * 32) return;
  int c = t & 31, n = (t >> 5) & (NPTS - 1), b = t >> 15;
  float v = (c < 3) ? x[((size_t)b * 3 + c) * NPTS + n] : 0.f;
  xb0[t] = f2bf(v);
}

// w (O,Cin) f32 -> wb (O,Kp) bf16 zero padded
__global__ void cast_w_kernel(const float* __restrict__ w, unsigned short* __restrict__ wb,
                              int O, int Cin, int Kp) {
  int t = blockIdx.x * blockDim.x + threadIdx.x;
  if (t >= O * Kp) return;
  int k = t % Kp, o = t / Kp;
  wb[t] = f2bf(k < Cin ? w[(size_t)o * Cin + k] : 0.f);
}

// per-point squared norm over C channels of a bf16 slice
__global__ void xx_kernel(const unsigned short* __restrict__ fb, int rstride, int off, int C,
                          float* __restrict__ xx) {
  int t = blockIdx.x * blockDim.x + threadIdx.x;
  if (t >= BATCH * NPTS) return;
  const unsigned short* p = fb + (size_t)t * rstride + off;
  float s = 0.f;
  for (int c = 0; c < C; ++c) { float v = bf2f(p[c]); s += v * v; }
  xx[t] = s;
}

// ------------------------- kNN: gram (WMMA) + top-k ------------------------
// grid: B*64 blocks (16-row tiles), block: 512 threads (16 waves)
// dyn LDS: 16*1024 floats (64 KB).  NKC = Kp/32 K-chunks (compile-time).
template <int NKC>
__global__ void knn_kernel(const unsigned short* __restrict__ fb, int rstride, int off,
                           const float* __restrict__ xx, int* __restrict__ idxout) {
  extern __shared__ float dist[];                 // [16][1024]
  int tid = threadIdx.x, lane = tid & 31, wave = tid >> 5;
  int b = blockIdx.x >> 6, rt = blockIdx.x & 63;
  int rowbase = rt * 16;
  const unsigned short* fbb = fb + (size_t)b * NPTS * rstride + off;
  const float* xxb = xx + b * NPTS;

  v16bf afr[NKC];
#pragma unroll
  for (int kc = 0; kc < NKC; ++kc)
    afr[kc] = load_frag_a(fbb + (size_t)rowbase * rstride + kc * 32, rstride, lane);

  int n = lane & 15, h = lane >> 4;
  float xxr[8];
#pragma unroll
  for (int r = 0; r < 8; ++r) xxr[r] = xxb[rowbase + r + 8 * h];

  for (int ct = wave; ct < 64; ct += 16) {        // uniform per wave
    v8f acc = {0.f, 0.f, 0.f, 0.f, 0.f, 0.f, 0.f, 0.f};
#pragma unroll
    for (int kc = 0; kc < NKC; ++kc) {
      v16bf bf = load_frag_b(fbb + (size_t)(ct * 16) * rstride + kc * 32, rstride, lane);
      acc = wmma_bf16(afr[kc], bf, acc);
    }
    float xxc = xxb[ct * 16 + n];
#pragma unroll
    for (int r = 0; r < 8; ++r) {
      int m = r + 8 * h;                          // D: M = r + 8h, N = lane&15
      dist[m * NPTS + ct * 16 + n] = 2.f * acc[r] - xxr[r] - xxc;
    }
  }
  __syncthreads();

  // top-20 per row: wave w owns row w; iterative argmax extraction
  int row = wave;
  float d[32];
#pragma unroll
  for (int i = 0; i < 32; ++i) d[i] = dist[row * NPTS + lane + 32 * i];
  int* op = idxout + ((size_t)(b * NPTS + rowbase + row)) * KNN;
#pragma unroll 1
  for (int it = 0; it < KNN; ++it) {
    float best = d[0];
    int bslot = 0;
#pragma unroll
    for (int i = 1; i < 32; ++i)
      if (d[i] > best) { best = d[i]; bslot = i; }
    int bcol = bslot * 32 + lane;
#pragma unroll
    for (int m = 16; m; m >>= 1) {
      float ov = __shfl_xor(best, m, 32);
      int   oc = __shfl_xor(bcol, m, 32);
      if (ov > best || (ov == best && oc < bcol)) { best = ov; bcol = oc; }
    }
    if (lane == 0) op[it] = bcol;
    if ((bcol & 31) == lane) d[bcol >> 5] = -3.0e38f;
  }
}

// ------------------------- EdgeConv (gather + WMMA + max-k) ----------------
// grid: B*256 blocks (4-point blocks -> 80 edges), block: 512 threads
// dyn LDS: Kp*80 bf16 (G, column-major) + O*4 uint (max keys)
// NKC = Kp/32 (compile-time), Kp = 2C rounded up to 32.
template <int NKC>
__global__ void edgeconv_kernel(const unsigned short* __restrict__ fb, int rstride, int off,
                                int C, const int* __restrict__ idxg,
                                const unsigned short* __restrict__ wb,
                                const float* __restrict__ sv, const float* __restrict__ bv,
                                int O, unsigned short* __restrict__ outfb, int outoff,
                                int outstride) {
  constexpr int Kp = NKC * 32;
  extern __shared__ char smem[];
  unsigned short* Gs = (unsigned short*)smem;               // [80][Kp] col-major
  unsigned* okey = (unsigned*)(smem + (size_t)Kp * 80 * 2); // [O][4]
  int tid = threadIdx.x, lane = tid & 31, wave = tid >> 5;
  int b = blockIdx.x >> 8, pb = blockIdx.x & 255;
  int nbase = pb * 4;
  const unsigned short* fbb = fb + (size_t)b * NPTS * rstride + off;

  for (int t = tid; t < O * 4; t += blockDim.x) okey[t] = 0u;

  int twoC = 2 * C;
  for (int t = tid; t < 80 * Kp; t += blockDim.x) {
    int e = t / Kp, c = t - e * Kp;
    int p = e / KNN, j = e - p * KNN;
    int n = nbase + p;
    float val = 0.f;
    if (c < twoC) {
      if (c < C) {
        int nbr = idxg[(size_t)(b * NPTS + n) * KNN + j];
        val = bf2f(fbb[(size_t)nbr * rstride + c]) - bf2f(fbb[(size_t)n * rstride + c]);
      } else {
        val = bf2f(fbb[(size_t)n * rstride + (c - C)]);
      }
    }
    Gs[(size_t)e * Kp + c] = f2bf(val);
  }
  __syncthreads();

  int Mt = O >> 4;
  int njobs = Mt * 5;                                      // 5 col tiles of 16 edges
  int nwaves = blockDim.x >> 5;
  for (int job = wave; job < njobs; job += nwaves) {       // uniform per wave
    int mt = job / 5, et = job - mt * 5;
    v8f acc = {0.f, 0.f, 0.f, 0.f, 0.f, 0.f, 0.f, 0.f};
#pragma unroll
    for (int kc = 0; kc < NKC; ++kc) {
      v16bf a = load_frag_a(wb + (size_t)(mt * 16) * Kp + kc * 32, Kp, lane);
      v16bf g = load_frag_b(Gs + (size_t)(et * 16) * Kp + kc * 32, Kp, lane);
      acc = wmma_bf16(a, g, acc);
    }
    int n = lane & 15, h = lane >> 4;
    int e = et * 16 + n;
    int p = e / KNN;
#pragma unroll
    for (int r = 0; r < 8; ++r) {
      int o = mt * 16 + r + 8 * h;
      float y = acc[r] * sv[o] + bv[o];
      y = (y >= 0.f) ? y : 0.2f * y;
      atomicMax(&okey[o * 4 + p], fkey(y));                // ds_max_u32
    }
  }
  __syncthreads();

  for (int t = tid; t < O * 4; t += blockDim.x) {
    int o = t >> 2, p = t & 3;
    float v = finv(okey[t]);
    outfb[(size_t)(b * NPTS + nbase + p) * outstride + outoff + o] = f2bf(v);
  }
}

// ------------------------- final 512x512 conv ------------------------------
// grid: 4096 blocks x 256 threads (8 waves); one 16x16 tile per wave
__global__ void finalconv_kernel(const unsigned short* __restrict__ xcb,
                                 const unsigned short* __restrict__ wb,
                                 const float* __restrict__ sv, const float* __restrict__ bv,
                                 float* __restrict__ xinv) {
  int tid = threadIdx.x, lane = tid & 31, wave = tid >> 5;
  int job = blockIdx.x * 8 + wave;                 // 16 * 32 * 64 jobs
  int b = job >> 11;
  int rem = job & 2047;
  int mt = rem >> 6, nt = rem & 63;
  const unsigned short* xb = xcb + ((size_t)b * NPTS + nt * 16) * 512;
  v8f acc = {0.f, 0.f, 0.f, 0.f, 0.f, 0.f, 0.f, 0.f};
#pragma unroll 4
  for (int kc = 0; kc < 16; ++kc) {
    v16bf a = load_frag_a(wb + (size_t)(mt * 16) * 512 + kc * 32, 512, lane);
    v16bf g = load_frag_b(xb + kc * 32, 512, lane);
    acc = wmma_bf16(a, g, acc);
  }
  int n = lane & 15, h = lane >> 4;
  int col = nt * 16 + n;
#pragma unroll
  for (int r = 0; r < 8; ++r) {
    int o = mt * 16 + r + 8 * h;
    float y = acc[r] * sv[o] + bv[o];
    y = (y >= 0.f) ? y : 0.2f * y;
    xinv[((size_t)b * 512 + o) * NPTS + col] = y;
  }
}

// ------------------------- pooling -----------------------------------------
// wave per (b,o) row: max + mean over N
__global__ void pool_kernel(const float* __restrict__ xinv, float* __restrict__ xequ) {
  int tid = threadIdx.x, lane = tid & 31, wave = tid >> 5;
  int row = blockIdx.x * 8 + wave;                 // B*512 rows
  int b = row >> 9, o = row & 511;
  const float* p = xinv + ((size_t)b * 512 + o) * NPTS;
  float mx = -3.0e38f, sm = 0.f;
#pragma unroll
  for (int i = 0; i < 32; ++i) {
    float v = p[lane + 32 * i];
    mx = fmaxf(mx, v);
    sm += v;
  }
#pragma unroll
  for (int m = 16; m; m >>= 1) {
    mx = fmaxf(mx, __shfl_xor(mx, m, 32));
    sm += __shfl_xor(sm, m, 32);
  }
  if (lane == 0) {
    xequ[b * 1024 + o]       = mx;
    xequ[b * 1024 + 512 + o] = sm * (1.f / 1024.f);
  }
}

// ---------------------------------------------------------------------------
extern "C" void kernel_launch(void* const* d_in, const int* in_sizes, int n_in,
                              void* d_out, int out_size, void* d_ws, size_t ws_size,
                              hipStream_t stream) {
  (void)in_sizes; (void)n_in; (void)out_size; (void)ws_size;
  const float* x = (const float*)d_in[0];
  const float* w[5], *sv[5], *bv[5];
  for (int i = 0; i < 5; ++i) {
    w[i]  = (const float*)d_in[1 + 3 * i];
    sv[i] = (const float*)d_in[2 + 3 * i];
    bv[i] = (const float*)d_in[3 + 3 * i];
  }
  float* out = (float*)d_out;
  float* xequ = out;                 // (16,1024)
  float* xinv = out + 16 * 1024;     // (16,512,1024)

  // workspace carve-up
  char* ws = (char*)d_ws;
  size_t off = 0;
  auto carve = [&](size_t bytes) { void* p = ws + off; off += (bytes + 255) & ~(size_t)255; return p; };
  unsigned short* xb0 = (unsigned short*)carve((size_t)BATCH * NPTS * 32 * 2);
  unsigned short* xcb = (unsigned short*)carve((size_t)BATCH * NPTS * 512 * 2);
  float*          xxb = (float*)carve((size_t)BATCH * NPTS * 4);
  int*            idx = (int*)carve((size_t)BATCH * NPTS * KNN * 4);
  unsigned short* wb[5];
  const int wO[5]  = {64, 64, 128, 256, 512};
  const int wC[5]  = {6, 128, 128, 256, 512};
  const int wKp[5] = {32, 128, 128, 256, 512};
  for (int i = 0; i < 5; ++i)
    wb[i] = (unsigned short*)carve((size_t)wO[i] * wKp[i] * 2);

  // casts
  cast_x0_kernel<<<(BATCH * NPTS * 32 + 255) / 256, 256, 0, stream>>>(x, xb0);
  for (int i = 0; i < 5; ++i)
    cast_w_kernel<<<(wO[i] * wKp[i] + 255) / 256, 256, 0, stream>>>(w[i], wb[i], wO[i], wC[i], wKp[i]);

  const int knnLds = 16 * NPTS * 4;

  // ---- layer 1: knn on raw x (C=3, Kp=32), edgeconv -> xcb[:,:,0:64)
  xx_kernel<<<(BATCH * NPTS + 255) / 256, 256, 0, stream>>>(xb0, 32, 0, 3, xxb);
  knn_kernel<1><<<BATCH * 64, 512, knnLds, stream>>>(xb0, 32, 0, xxb, idx);
  edgeconv_kernel<1><<<BATCH * 256, 512, 32 * 80 * 2 + 64 * 4 * 4, stream>>>(
      xb0, 32, 0, 3, idx, wb[0], sv[0], bv[0], 64, xcb, 0, 512);

  // ---- layer 2: knn on x1 (C=64), edgeconv -> xcb[:,:,64:128)
  xx_kernel<<<(BATCH * NPTS + 255) / 256, 256, 0, stream>>>(xcb, 512, 0, 64, xxb);
  knn_kernel<2><<<BATCH * 64, 512, knnLds, stream>>>(xcb, 512, 0, xxb, idx);
  edgeconv_kernel<4><<<BATCH * 256, 512, 128 * 80 * 2 + 64 * 4 * 4, stream>>>(
      xcb, 512, 0, 64, idx, wb[1], sv[1], bv[1], 64, xcb, 64, 512);

  // ---- layer 3: knn on x2 (C=64), edgeconv -> xcb[:,:,128:256)
  xx_kernel<<<(BATCH * NPTS + 255) / 256, 256, 0, stream>>>(xcb, 512, 64, 64, xxb);
  knn_kernel<2><<<BATCH * 64, 512, knnLds, stream>>>(xcb, 512, 64, xxb, idx);
  edgeconv_kernel<4><<<BATCH * 256, 512, 128 * 80 * 2 + 128 * 4 * 4, stream>>>(
      xcb, 512, 64, 64, idx, wb[2], sv[2], bv[2], 128, xcb, 128, 512);

  // ---- layer 4: knn on x3 (C=128), edgeconv -> xcb[:,:,256:512)
  xx_kernel<<<(BATCH * NPTS + 255) / 256, 256, 0, stream>>>(xcb, 512, 128, 128, xxb);
  knn_kernel<4><<<BATCH * 64, 512, knnLds, stream>>>(xcb, 512, 128, xxb, idx);
  edgeconv_kernel<8><<<BATCH * 256, 512, 256 * 80 * 2 + 256 * 4 * 4, stream>>>(
      xcb, 512, 128, 128, idx, wb[3], sv[3], bv[3], 256, xcb, 256, 512);

  // ---- final conv + pooling
  finalconv_kernel<<<4096, 256, 0, stream>>>(xcb, wb[4], sv[4], bv[4], xinv);
  pool_kernel<<<1024, 256, 0, stream>>>(xinv, xequ);
}
